// HeteroGNN_31610959298705
// MI455X (gfx1250) — matrix operand order repaired and checked by reading, not compiled, not run on hardware
//
#include <hip/hip_runtime.h>
#include <hip/hip_bf16.h>

// CDNA5 (gfx1250) hetero-GraphSAGE:
//   wave32, WMMA f32_16x16x32_f16 GEMMs (f16 fragments, f32 accumulation),
//   L2-resident scatter-mean via f32 global atomics.

typedef _Float16 v16h __attribute__((ext_vector_type(16)));
typedef float    v8f  __attribute__((ext_vector_type(8)));

#define HDIM 128

struct SegDesc {
    const float* A;    // M x K row-major activations (f32)
    const float* cnt;  // M counts (mean denominator) or nullptr
    const float* W;    // K x 128 row-major weights (f32)
    int K;             // 64 or 128 (multiple of 32)
};

// ---------------------------------------------------------------------------
// C[M x 128] = sum_seg ( (A_seg * scale_seg) @ W_seg ) + bias
//   scale = 1/max(cnt,1) when cnt != nullptr (fused SAGE mean)
//   accumulate: C += result; leaky: fused leaky-ReLU(0.01) epilogue.
// 256 threads = 8 waves; wave w owns rows [m0, m0+16) x all 128 cols.
// ---------------------------------------------------------------------------
__global__ __launch_bounds__(256)
void sage_gemm128(SegDesc s0, SegDesc s1, int nseg, int M,
                  const float* bias, float* __restrict__ C,
                  int accumulate, int leaky)
{
    // B fragments pre-packed in WMMA layout: [ntile(8)][kstep(4)][lane(32)][16 f16]
    __shared__ __align__(32) _Float16 ldsB[8 * 4 * 32 * 16];   // 32 KB

    const int tid  = threadIdx.x;
    const int wave = tid >> 5;
    const int lane = tid & 31;
    const int m0   = blockIdx.x * 128 + wave * 16;
    const bool active = (m0 < M);          // M is a multiple of 16 -> full tiles
    const int rrow = m0 + (lane & 15);     // A/C row this lane touches

    v8f acc[8] = {};                       // 8 N-tiles x 8 f32 = 16x128 strip

    for (int s = 0; s < nseg; ++s) {
        SegDesc sd = (s == 0) ? s0 : s1;
        const int ksteps = sd.K >> 5;      // K / 32

        __syncthreads();                   // previous segment's B consumed
        // ---- pack W (K x 128 f32) into WMMA B fragments (f16) in LDS ----
        const int items = 8 * ksteps * 32; // (ntile, kstep, lane) triples
        for (int it = tid; it < items; it += 256) {
            int l  = it & 31;
            int g  = it >> 5;
            int ks = g % ksteps;
            int nt = g / ksteps;
            int n     = nt * 16 + (l & 15);
            int kbase = ks * 32 + (l >> 4) * 16;  // lanes 16-31 -> K+16..31
            _Float16* dp = &ldsB[(size_t)((nt * 4 + ks) * 32 + l) * 16];
#pragma unroll
            for (int v = 0; v < 8; ++v) {
                int k = kbase + 2 * v;
                dp[2 * v]     = (_Float16)sd.W[(size_t)k * HDIM + n];
                dp[2 * v + 1] = (_Float16)sd.W[(size_t)(k + 1) * HDIM + n];
            }
        }
        __syncthreads();

        if (active) {
            float sc = 1.0f;
            if (sd.cnt) sc = 1.0f / fmaxf(sd.cnt[rrow], 1.0f);  // SAGE mean
            const float* arow = sd.A + (size_t)rrow * sd.K;
            const int kh = (lane >> 4) * 16;  // A layout K offset per half-wave

            for (int ks = 0; ks < ksteps; ++ks) {
                const int k0 = ks * 32;
                v16h afrag;
#pragma unroll
                for (int v = 0; v < 8; ++v) {
                    // A 16x32 layout (contiguous, matches B packing):
                    //  lanes 0-15:  v0-7 -> K0..15
                    //  lanes16-31:  v0-7 -> K16..31
                    int kk = 2 * v + kh;
                    float2 p = *(const float2*)(arow + k0 + kk);
                    afrag[2 * v]     = (_Float16)(p.x * sc);
                    afrag[2 * v + 1] = (_Float16)(p.y * sc);
                }
#pragma unroll
                for (int nt = 0; nt < 8; ++nt) {
                    v16h bfrag = *(const v16h*)
                        &ldsB[(size_t)((nt * 4 + ks) * 32 + lane) * 16];
                    acc[nt] = __builtin_amdgcn_wmma_f32_16x16x32_f16(
                        false, afrag, false, bfrag, (short)0, acc[nt],
                        false, false);
                }
            }
        }
    }

    if (active) {
        const int n  = lane & 15;          // C layout: N = lane&15
        const int mh = lane >> 4;          // lanes 16-31 -> M rows 8..15
#pragma unroll
        for (int nt = 0; nt < 8; ++nt) {
            int col  = nt * 16 + n;
            float bb = bias ? bias[col] : 0.0f;
#pragma unroll
            for (int v = 0; v < 8; ++v) {
                size_t idx = (size_t)(m0 + mh * 8 + v) * HDIM + col;
                float val = acc[nt][v] + bb;
                if (accumulate) val += C[idx];
                if (leaky) val = (val >= 0.0f) ? val : 0.01f * val;
                C[idx] = val;
            }
        }
    }
}

// ---------------------------------------------------------------------------
// Output head: C[M x 16] = A[M x 128] @ W[128 x 16] + bias (single N-tile WMMA)
// ---------------------------------------------------------------------------
__global__ __launch_bounds__(256)
void out_head_kernel(const float* __restrict__ A, const float* __restrict__ W,
                     const float* __restrict__ bias, float* __restrict__ C, int M)
{
    __shared__ __align__(32) _Float16 ldsB[4 * 32 * 16];   // 4 KB

    const int tid  = threadIdx.x;
    const int wave = tid >> 5;
    const int lane = tid & 31;
    const int m0   = blockIdx.x * 128 + wave * 16;

    for (int it = tid; it < 4 * 32; it += 256) {
        int l  = it & 31;
        int ks = it >> 5;
        int n     = l & 15;
        int kbase = ks * 32 + (l >> 4) * 16;
        _Float16* dp = &ldsB[(size_t)(ks * 32 + l) * 16];
#pragma unroll
        for (int v = 0; v < 8; ++v) {
            int k = kbase + 2 * v;
            dp[2 * v]     = (_Float16)W[(size_t)k * 16 + n];
            dp[2 * v + 1] = (_Float16)W[(size_t)(k + 1) * 16 + n];
        }
    }
    __syncthreads();
    if (m0 >= M) return;

    const int rrow = m0 + (lane & 15);
    const float* arow = A + (size_t)rrow * HDIM;
    const int kh = (lane >> 4) * 16;
    v8f acc = {};
    for (int ks = 0; ks < 4; ++ks) {
        const int k0 = ks * 32;
        v16h afrag;
#pragma unroll
        for (int v = 0; v < 8; ++v) {
            int kk = 2 * v + kh;
            float2 p = *(const float2*)(arow + k0 + kk);
            afrag[2 * v]     = (_Float16)p.x;
            afrag[2 * v + 1] = (_Float16)p.y;
        }
        v16h bfrag = *(const v16h*)&ldsB[(size_t)(ks * 32 + lane) * 16];
        acc = __builtin_amdgcn_wmma_f32_16x16x32_f16(
            false, afrag, false, bfrag, (short)0, acc, false, false);
    }
    const int n  = lane & 15;
    const int mh = lane >> 4;
    float bb = bias[n];
#pragma unroll
    for (int v = 0; v < 8; ++v)
        C[(size_t)(m0 + mh * 8 + v) * 16 + n] = acc[v] + bb;
}

// ---------------------------------------------------------------------------
// One wave per edge: agg[dst] += h[src] (128 f32), cnt[dst] += 1.
// h rows and agg rows are L2-resident on MI455X (192 MB L2).
// ---------------------------------------------------------------------------
__global__ __launch_bounds__(256)
void scatter_add_kernel(const float* __restrict__ h, const int* __restrict__ src,
                        const int* __restrict__ dst, float* __restrict__ agg,
                        float* __restrict__ cnt, int E)
{
    int e = blockIdx.x * 8 + (threadIdx.x >> 5);
    if (e >= E) return;
    int lane = threadIdx.x & 31;
    int s = src[e];
    int d = dst[e];
    const float4 v = *(const float4*)(h + (size_t)s * HDIM + lane * 4);
    float* ap = agg + (size_t)d * HDIM + lane * 4;
    atomicAdd(ap + 0, v.x);
    atomicAdd(ap + 1, v.y);
    atomicAdd(ap + 2, v.z);
    atomicAdd(ap + 3, v.w);
    if (lane == 0) atomicAdd(cnt + d, 1.0f);
}

__global__ void zero_kernel(float* __restrict__ p, size_t n)
{
    size_t i  = (size_t)blockIdx.x * blockDim.x + threadIdx.x;
    size_t st = (size_t)gridDim.x * blockDim.x;
    for (; i < n; i += st) p[i] = 0.0f;
}

// ---------------------------------------------------------------------------
extern "C" void kernel_launch(void* const* d_in, const int* in_sizes, int n_in,
                              void* d_out, int out_size, void* d_ws, size_t ws_size,
                              hipStream_t stream)
{
    (void)n_in; (void)out_size; (void)ws_size;

    const float* x_user    = (const float*)d_in[0];
    const float* x_item    = (const float*)d_in[1];
    const int*   src_rates = (const int*)d_in[2];
    const int*   dst_rates = (const int*)d_in[3];
    const int*   src_rated = (const int*)d_in[4];
    const int*   dst_rated = (const int*)d_in[5];
    const int*   src_fol   = (const int*)d_in[6];
    const int*   dst_fol   = (const int*)d_in[7];
    const float* W_in_user = (const float*)d_in[8];
    const float* b_in_user = (const float*)d_in[9];
    const float* W_in_item = (const float*)d_in[10];
    const float* b_in_item = (const float*)d_in[11];
    const float* Wl_rates  = (const float*)d_in[12];
    const float* bl_rates  = (const float*)d_in[13];
    const float* Wr_rates  = (const float*)d_in[14];
    const float* Wl_rated  = (const float*)d_in[15];
    const float* bl_rated  = (const float*)d_in[16];
    const float* Wr_rated  = (const float*)d_in[17];
    const float* Wl_fol    = (const float*)d_in[18];
    const float* bl_fol    = (const float*)d_in[19];
    const float* Wr_fol    = (const float*)d_in[20];
    const float* W_out     = (const float*)d_in[21];
    const float* b_out     = (const float*)d_in[22];

    const int NU = in_sizes[0] / 64;    // 200000
    const int NI = in_sizes[1] / 128;   // 100000
    const int E  = in_sizes[2];         // 1000000

    // Workspace layout (f32): hu | hi | hu2 | hi2 | aggU cntU | aggI cntI
    float* ws  = (float*)d_ws;
    size_t off = 0;
    float* hu   = ws + off; off += (size_t)NU * HDIM;
    float* hi   = ws + off; off += (size_t)NI * HDIM;
    float* hu2  = ws + off; off += (size_t)NU * HDIM;
    float* hi2  = ws + off; off += (size_t)NI * HDIM;
    float* aggU = ws + off; off += (size_t)NU * HDIM;
    float* cntU = ws + off; off += (size_t)NU;            // contiguous w/ aggU
    float* aggI = ws + off; off += (size_t)NI * HDIM;
    float* cntI = ws + off; off += (size_t)NI;            // contiguous w/ aggI

    const dim3 blk(256);
    const int gU = (NU + 127) / 128;
    const int gI = (NI + 127) / 128;
    const int gE = (E + 7) / 8;
    const size_t HH = (size_t)HDIM * HDIM;

    SegDesc d0, d1;

    // Input projections
    d0 = { x_user, nullptr, W_in_user, 64 };
    sage_gemm128<<<gU, blk, 0, stream>>>(d0, d0, 1, NU, b_in_user, hu, 0, 0);
    d0 = { x_item, nullptr, W_in_item, 128 };
    sage_gemm128<<<gI, blk, 0, stream>>>(d0, d0, 1, NI, b_in_item, hi, 0, 0);

    float* curU = hu; float* curI = hi; float* nxtU = hu2; float* nxtI = hi2;

    for (int l = 0; l < 2; ++l) {
        const int lk = (l == 0) ? 1 : 0;   // leaky between layers only

        // user->item ('rates'): hi_new = mean@Wl + bl + hi@Wr
        zero_kernel<<<2048, blk, 0, stream>>>(aggI, (size_t)NI * HDIM + NI);
        scatter_add_kernel<<<gE, blk, 0, stream>>>(curU, src_rates, dst_rates,
                                                   aggI, cntI, E);
        d0 = { aggI, cntI, Wl_rates + l * HH, 128 };
        d1 = { curI, nullptr, Wr_rates + l * HH, 128 };
        sage_gemm128<<<gI, blk, 0, stream>>>(d0, d1, 2, NI, bl_rates + l * HDIM,
                                             nxtI, 0, lk);

        // item->user ('rated_by')
        zero_kernel<<<2048, blk, 0, stream>>>(aggU, (size_t)NU * HDIM + NU);
        scatter_add_kernel<<<gE, blk, 0, stream>>>(curI, src_rated, dst_rated,
                                                   aggU, cntU, E);
        d0 = { aggU, cntU, Wl_rated + l * HH, 128 };
        d1 = { curU, nullptr, Wr_rated + l * HH, 128 };
        sage_gemm128<<<gU, blk, 0, stream>>>(d0, d1, 2, NU, bl_rated + l * HDIM,
                                             nxtU, 0, 0);

        // user->user ('follows'), HeteroConv sum: accumulate into nxtU
        zero_kernel<<<2048, blk, 0, stream>>>(aggU, (size_t)NU * HDIM + NU);
        scatter_add_kernel<<<gE, blk, 0, stream>>>(curU, src_fol, dst_fol,
                                                   aggU, cntU, E);
        d0 = { aggU, cntU, Wl_fol + l * HH, 128 };
        d1 = { curU, nullptr, Wr_fol + l * HH, 128 };
        sage_gemm128<<<gU, blk, 0, stream>>>(d0, d1, 2, NU, bl_fol + l * HDIM,
                                             nxtU, 1, lk);

        float* t = curU; curU = nxtU; nxtU = t;
        t = curI; curI = nxtI; nxtI = t;
    }

    // Output head: hu @ W_out + b_out  -> d_out [NU x 16] f32
    out_head_kernel<<<gU, blk, 0, stream>>>(curU, W_out, b_out, (float*)d_out, NU);
}
